// HebbianBlock_54176717472174
// MI455X (gfx1250) — matrix operand order repaired and checked by reading, not compiled, not run on hardware
//
#include <hip/hip_runtime.h>
#include <hip/hip_bf16.h>
#include <math.h>

// Problem constants (from reference: B=4, T=4096, D=1024, H=8, C=64)
#define BB 4
#define TT 4096
#define DD 1024
#define HH 8
#define DH 128            // head dim d = D/H
#define CC 64             // chunk length
#define NN (TT / CC)      // 64 chunks
#define BT (BB * TT)      // 16384 rows

typedef float v2f  __attribute__((ext_vector_type(2)));
typedef float v8f  __attribute__((ext_vector_type(8)));
typedef __bf16 v16bf __attribute__((ext_vector_type(16)));

static __device__ __forceinline__ v8f vzero8() {
  v8f z;
#pragma unroll
  for (int i = 0; i < 8; ++i) z[i] = 0.f;
  return z;
}

static __device__ __forceinline__ v8f wmma_f32_k4(v2f a, v2f b, v8f c) {
  // D = A(16x4 f32) * B(4x16 f32) + C ; fp32 exact path for the recurrence
  return __builtin_amdgcn_wmma_f32_16x16x4_f32(false, a, false, b, (short)0, c,
                                               false, false);
}
static __device__ __forceinline__ v8f wmma_bf16_k32(v16bf a, v16bf b, v8f c) {
  // D = A(16x32 bf16) * B(32x16 bf16) + C(f32) ; high-throughput projection path
  return __builtin_amdgcn_wmma_f32_16x16x32_bf16(false, a, false, b, (short)0, c,
                                                 false, false);
}

// ---------------------------------------------------------------------------
// Kernel 1/5: Y[m,n] = sum_k X[m,k]*W[n,k] (+ res[m,n]).  M=BT, N=K=DD.
// 128x128 C-tile per 256-thread WG (8 waves, each 32x64 = 2x4 WMMA tiles).
// K staged in LDS as bf16 in 64-wide slabs (hardware cvt at staging time);
// bf16 WMMA with fp32 accumulate. Next slab prefetched via global_prefetch.
// ---------------------------------------------------------------------------
__global__ __launch_bounds__(256) void gemm_nt_bf16_kernel(
    const float* __restrict__ X, const float* __restrict__ W,
    const float* __restrict__ res, float* __restrict__ Y) {
  __shared__ __bf16 sX[128][72];   // 18 KB
  __shared__ __bf16 sW[128][72];   // 18 KB
  const int tid  = threadIdx.x;
  const int wave = tid >> 5, lane = tid & 31;
  const int lh = lane >> 4, ll = lane & 15;
  const int mw = wave >> 1;        // 0..3 -> 32-row strip
  const int nw = wave & 1;         // 0..1 -> 64-col half
  const int m0 = blockIdx.y * 128;
  const int n0 = blockIdx.x * 128;

  v8f acc[2][4];
#pragma unroll
  for (int mt = 0; mt < 2; ++mt)
#pragma unroll
    for (int nt = 0; nt < 4; ++nt) acc[mt][nt] = vzero8();

  for (int k0 = 0; k0 < DD; k0 += 64) {
    for (int i = tid; i < 128 * 64; i += 256) {
      int r = i >> 6, c = i & 63;
      sX[r][c] = (__bf16)X[(size_t)(m0 + r) * DD + k0 + c];
      sW[r][c] = (__bf16)W[(size_t)(n0 + r) * DD + k0 + c];
      if ((c == 0) && (k0 + 64 < DD)) {     // prefetch next K-slab rows
        __builtin_prefetch(&X[(size_t)(m0 + r) * DD + k0 + 64], 0, 3);
        __builtin_prefetch(&W[(size_t)(n0 + r) * DD + k0 + 64], 0, 3);
      }
    }
    __syncthreads();
#pragma unroll
    for (int kt = 0; kt < 2; ++kt) {
      const int kb = kt * 32;
      // A fragments: 16x32 bf16 (ISA layout: k = 16*(e>=8) + 8*hi + (e&7))
      v16bf a[2];
#pragma unroll
      for (int mt = 0; mt < 2; ++mt) {
        const int am = mw * 32 + mt * 16 + ll;
#pragma unroll
        for (int e = 0; e < 16; ++e) {
          int k = kb + ((e >> 3) << 4) + (lh << 3) + (e & 7);
          a[mt][e] = sX[am][k];
        }
      }
#pragma unroll
      for (int nt = 0; nt < 4; ++nt) {
        v16bf b;
        const int bn = nw * 64 + nt * 16 + ll;
        // B layout (ISA): k = 16*(e>=8) + 8*hi + (e&7)
#pragma unroll
        for (int e = 0; e < 16; ++e) b[e] = sW[bn][kb + ((e >> 3) << 4) + (lh << 3) + (e & 7)];
#pragma unroll
        for (int mt = 0; mt < 2; ++mt)
          acc[mt][nt] = wmma_bf16_k32(a[mt], b, acc[mt][nt]);
      }
    }
    __syncthreads();
  }
#pragma unroll
  for (int mt = 0; mt < 2; ++mt) {
#pragma unroll
    for (int nt = 0; nt < 4; ++nt) {
      const int col = n0 + nw * 64 + nt * 16 + ll;
#pragma unroll
      for (int r = 0; r < 8; ++r) {
        const int row = m0 + mw * 32 + mt * 16 + r + (lh << 3);
        float vv = acc[mt][nt][r];
        if (res) vv += res[(size_t)row * DD + col];
        Y[(size_t)row * DD + col] = vv;
      }
    }
  }
}

// ---------------------------------------------------------------------------
// Kernel 2: per-row head-normalization rk = xh/||xh|| and beta = sigmoid(x@Wb^T)
// One workgroup per row; wave h (wave32) owns head h.
// ---------------------------------------------------------------------------
__global__ __launch_bounds__(256) void rk_beta_kernel(
    const float* __restrict__ X, const float* __restrict__ Wb,
    float* __restrict__ rk, float* __restrict__ beta) {
  const int t = blockIdx.x;                 // global row in [0, BT)
  const int h = threadIdx.x >> 5;
  const int lane = threadIdx.x & 31;
  const float* xr = X + (size_t)t * DD;

  float vals[4]; float ss = 0.f;
#pragma unroll
  for (int i = 0; i < 4; ++i) {
    float v = xr[h * DH + i * 32 + lane];
    vals[i] = v; ss += v * v;
  }
#pragma unroll
  for (int off = 16; off; off >>= 1) ss += __shfl_xor(ss, off, 32);
  const float inv = 1.0f / fmaxf(sqrtf(ss), 1e-12f);
#pragma unroll
  for (int i = 0; i < 4; ++i)
    rk[(size_t)t * DD + h * DH + i * 32 + lane] = vals[i] * inv;

  float acc = 0.f;
  for (int i = 0; i < DD / 32; ++i) {
    int j = i * 32 + lane;
    acc += xr[j] * Wb[(size_t)h * DD + j];
  }
#pragma unroll
  for (int off = 16; off; off >>= 1) acc += __shfl_xor(acc, off, 32);
  if (lane == 0) beta[(size_t)t * HH + h] = 1.0f / (1.0f + expf(-acc));
}

// ---------------------------------------------------------------------------
// Kernel 3: intra-chunk (fully parallel over B*H*N = 2048 chunks).
//   wkb = wk*beta ; A = -(wkb @ wk^T) .* gamma^(i-j) (strict lower)
//   UT transform; A += I; vbuf <- A @ (v*beta); wkcum <- A @ (wkb .* gamma^(i+1))
// fp32 WMMA 16x16x4 throughout.
// ---------------------------------------------------------------------------
__global__ __launch_bounds__(128) void chunk_ut_kernel(
    const float* __restrict__ rk, const float* __restrict__ beta,
    const float* __restrict__ decay, float* __restrict__ vbuf,
    float* __restrict__ wkcum) {
  __shared__ float sWkb[CC][DH + 4];   // 33.0 KB
  __shared__ float sA[CC][CC + 4];     // 17.0 KB
  __shared__ float sBeta[CC];
  __shared__ float sGpow[CC + 1];      // gamma^i

  const int idx = blockIdx.x;
  const int n = idx % NN;
  const int h = (idx / NN) % HH;
  const int b = idx / (NN * HH);
  const int tid = threadIdx.x, wave = tid >> 5, lane = tid & 31;
  const int lh = lane >> 4, ll = lane & 15;
  const size_t rowbase = (size_t)(b * TT + n * CC);
  const float gamma = 1.0f / (1.0f + expf(-decay[h]));

  if (tid <= CC) sGpow[tid] = powf(gamma, (float)tid);
  if (tid < CC)  sBeta[tid] = beta[(rowbase + tid) * HH + h];
  __syncthreads();

  // stage wkb = shifted(rk) * beta  (row r <- rk[rowbase+r-1], zero at t==0)
  for (int i = tid; i < CC * DH; i += 128) {
    int r = i >> 7, c = i & 127;
    float wv = 0.f;
    if (n * CC + r > 0) wv = rk[(rowbase + r - 1) * DD + h * DH + c];
    sWkb[r][c] = wv * sBeta[r];
  }
  __syncthreads();

  // ---- A = -(wkb @ wk^T) .* L  (strict lower) ----
  {
    const int m = wave * 16 + ll;
#pragma unroll
    for (int nt = 0; nt < 4; ++nt) {
      const int nj = nt * 16 + ll;
      const bool jz = (n * CC + nj == 0);
      const float* wkrow = rk + (rowbase + nj - 1) * DD + h * DH;
      v8f acc = vzero8();
      for (int k0 = 0; k0 < DH; k0 += 4) {
        v2f a, bf;
#pragma unroll
        for (int e = 0; e < 2; ++e) {
          int k = k0 + 2 * lh + e;
          a[e]  = sWkb[m][k];
          bf[e] = jz ? 0.f : wkrow[k];
        }
        acc = wmma_f32_k4(a, bf, acc);
      }
#pragma unroll
      for (int r = 0; r < 8; ++r) {
        int row = wave * 16 + r + (lh << 3);
        int col = nt * 16 + ll;
        sA[row][col] = (row > col) ? -acc[r] * sGpow[row - col] : 0.f;
      }
    }
  }
  __syncthreads();

  // ---- UT transform: row_i += row_i @ A, i = 1..C-1 (serial, in LDS) ----
  for (int i = 1; i < CC; ++i) {
    float s = 0.f;
    if (tid < i) {
      s = sA[i][tid];
      for (int j = tid + 1; j < i; ++j) s += sA[i][j] * sA[j][tid];
    }
    __syncthreads();
    if (tid < i) sA[i][tid] = s;
    __syncthreads();
  }
  if (tid < CC) sA[tid][tid] = 1.0f;
  __syncthreads();

  // cached A fragments along K=64 (16 k-steps x 2)
  const int m = wave * 16 + ll;
  float aA[32];
#pragma unroll
  for (int k0 = 0; k0 < CC; k0 += 4) {
    aA[k0 / 2 + 0] = sA[m][k0 + 2 * lh + 0];
    aA[k0 / 2 + 1] = sA[m][k0 + 2 * lh + 1];
  }

  // ---- wkcum = A @ (wkb .* gamma^(k+1)) ----
#pragma unroll
  for (int nt = 0; nt < 8; ++nt) {
    const int nn2 = nt * 16 + ll;
    v8f acc = vzero8();
    for (int k0 = 0; k0 < CC; k0 += 4) {
      v2f a, bf;
#pragma unroll
      for (int e = 0; e < 2; ++e) {
        int k = k0 + 2 * lh + e;
        a[e]  = aA[k0 / 2 + e];
        bf[e] = sWkb[k][nn2] * sGpow[k + 1];
      }
      acc = wmma_f32_k4(a, bf, acc);
    }
#pragma unroll
    for (int r = 0; r < 8; ++r) {
      int row = wave * 16 + r + (lh << 3);
      wkcum[(rowbase + row) * DD + h * DH + nn2] = acc[r];
    }
  }

  // ---- v' = A @ (v .* beta)  (in-place on vbuf: accumulate, sync, store) ----
  v8f vacc[8];
#pragma unroll
  for (int nt = 0; nt < 8; ++nt) {
    const int nn2 = nt * 16 + ll;
    v8f acc = vzero8();
    for (int k0 = 0; k0 < CC; k0 += 4) {
      v2f a, bf;
#pragma unroll
      for (int e = 0; e < 2; ++e) {
        int k = k0 + 2 * lh + e;
        a[e]  = aA[k0 / 2 + e];
        bf[e] = vbuf[(rowbase + k) * DD + h * DH + nn2] * sBeta[k];
      }
      acc = wmma_f32_k4(a, bf, acc);
    }
    vacc[nt] = acc;
  }
  __syncthreads();   // all reads of this chunk's vbuf rows done
#pragma unroll
  for (int nt = 0; nt < 8; ++nt) {
    const int nn2 = nt * 16 + ll;
#pragma unroll
    for (int r = 0; r < 8; ++r) {
      int row = wave * 16 + r + (lh << 3);
      vbuf[(rowbase + row) * DD + h * DH + nn2] = vacc[nt][r];
    }
  }
}

// ---------------------------------------------------------------------------
// Kernel 4: cross-chunk scan. State columns are independent -> split d=128
// into 4 groups of 32 (grid = B*H*4 = 128 WGs). S slab (128x32 f32) lives in
// LDS for the whole scan. fp32 WMMA 16x16x4.
// ---------------------------------------------------------------------------
__global__ __launch_bounds__(128) void scan_kernel(
    const float* __restrict__ rk, const float* __restrict__ vbuf,
    const float* __restrict__ wkcum, const float* __restrict__ decay,
    float* __restrict__ obuf) {
  __shared__ float sS[DH][36];         // 18.0 KB state slab
  __shared__ float sAttn[CC][CC + 4];  // 17.0 KB
  __shared__ float sVnew[CC][36];      //  9.0 KB
  __shared__ float sGpow[CC + 1];

  const int idx = blockIdx.x;
  const int g = idx & 3;
  const int h = (idx >> 2) % HH;
  const int b = idx / (4 * HH);
  const int tid = threadIdx.x, wave = tid >> 5, lane = tid & 31;
  const int lh = lane >> 4, ll = lane & 15;
  const int col0 = g * 32;
  const float gamma = 1.0f / (1.0f + expf(-decay[h]));
  const float gC = powf(gamma, (float)CC);

  if (tid <= CC) sGpow[tid] = powf(gamma, (float)tid);
  for (int i = tid; i < DH * 32; i += 128) sS[i >> 5][i & 31] = 0.f;
  __syncthreads();

  for (int n = 0; n < NN; ++n) {
    const size_t rowbase = (size_t)(b * TT + n * CC);
    const float* rkc = rk + rowbase * DD + h * DH;   // row i at rkc + i*DD

    // ---- attn = (rk @ wk^T) .* gamma^(i-j), causal incl diag ----
    {
      const int m = wave * 16 + ll;
      float aA[64];
#pragma unroll
      for (int k0 = 0; k0 < DH; k0 += 4) {
        aA[k0 / 2 + 0] = rkc[(size_t)m * DD + k0 + 2 * lh + 0];
        aA[k0 / 2 + 1] = rkc[(size_t)m * DD + k0 + 2 * lh + 1];
      }
#pragma unroll
      for (int nt = 0; nt < 4; ++nt) {
        const int j = nt * 16 + ll;
        const bool jz = (n * CC + j == 0);
        const float* wkrow = rkc + ((long)j - 1) * DD;
        v8f acc = vzero8();
        for (int k0 = 0; k0 < DH; k0 += 4) {
          v2f a, bf;
          a[0] = aA[k0 / 2]; a[1] = aA[k0 / 2 + 1];
          bf[0] = jz ? 0.f : wkrow[k0 + 2 * lh + 0];
          bf[1] = jz ? 0.f : wkrow[k0 + 2 * lh + 1];
          acc = wmma_f32_k4(a, bf, acc);
        }
#pragma unroll
        for (int r = 0; r < 8; ++r) {
          int row = wave * 16 + r + (lh << 3);
          int col = nt * 16 + ll;
          sAttn[row][col] = (row >= col) ? acc[r] * sGpow[row - col] : 0.f;
        }
      }
    }
    __syncthreads();

    // ---- v_new = v - wkcum @ S_cols ----
    {
      const int m = wave * 16 + ll;
      const float* wkcr = wkcum + rowbase * DD + h * DH;
      float aA[64];
#pragma unroll
      for (int k0 = 0; k0 < DH; k0 += 4) {
        aA[k0 / 2 + 0] = wkcr[(size_t)m * DD + k0 + 2 * lh + 0];
        aA[k0 / 2 + 1] = wkcr[(size_t)m * DD + k0 + 2 * lh + 1];
      }
#pragma unroll
      for (int nt = 0; nt < 2; ++nt) {
        v8f acc = vzero8();
        for (int k0 = 0; k0 < DH; k0 += 4) {
          v2f a, bf;
          int kk = k0 + 2 * lh;
          a[0] = aA[k0 / 2]; a[1] = aA[k0 / 2 + 1];
          bf[0] = sS[kk][nt * 16 + ll];
          bf[1] = sS[kk + 1][nt * 16 + ll];
          acc = wmma_f32_k4(a, bf, acc);
        }
#pragma unroll
        for (int r = 0; r < 8; ++r) {
          int row = wave * 16 + r + (lh << 3);
          int col = nt * 16 + ll;
          sVnew[row][col] =
              vbuf[(rowbase + row) * DD + h * DH + col0 + col] - acc[r];
        }
      }
    }
    __syncthreads();

    // ---- o = (rk .* gamma^(m+1)) @ S + attn @ v_new ----
    {
      const int m = wave * 16 + ll;
      const float dexpm = sGpow[m] * gamma;   // gamma^(m+1)
#pragma unroll
      for (int nt = 0; nt < 2; ++nt) {
        v8f acc = vzero8();
        for (int k0 = 0; k0 < DH; k0 += 4) {           // K=128 over S
          v2f a, bf;
          int kk = k0 + 2 * lh;
          a[0] = rkc[(size_t)m * DD + kk] * dexpm;
          a[1] = rkc[(size_t)m * DD + kk + 1] * dexpm;
          bf[0] = sS[kk][nt * 16 + ll];
          bf[1] = sS[kk + 1][nt * 16 + ll];
          acc = wmma_f32_k4(a, bf, acc);
        }
        for (int k0 = 0; k0 < CC; k0 += 4) {           // K=64 attn@v_new
          v2f a, bf;
          int kk = k0 + 2 * lh;
          a[0] = sAttn[m][kk]; a[1] = sAttn[m][kk + 1];
          bf[0] = sVnew[kk][nt * 16 + ll];
          bf[1] = sVnew[kk + 1][nt * 16 + ll];
          acc = wmma_f32_k4(a, bf, acc);
        }
#pragma unroll
        for (int r = 0; r < 8; ++r) {
          int row = wave * 16 + r + (lh << 3);
          int col = nt * 16 + ll;
          obuf[(rowbase + row) * DD + h * DH + col0 + col] = acc[r];
        }
      }
    }
    __syncthreads();   // all reads of sS done before update

    // ---- S = gamma^C * S + (wk .* gamma^(C-1-i))^T @ v_new ----
    {
#pragma unroll
      for (int mt = 0; mt < 2; ++mt) {
        const int p = wave * 32 + mt * 16 + ll;   // state row (feature idx)
#pragma unroll
        for (int nt = 0; nt < 2; ++nt) {
          v8f acc = vzero8();
          for (int k0 = 0; k0 < CC; k0 += 4) {
            v2f a, bf;
#pragma unroll
            for (int e = 0; e < 2; ++e) {
              int i = k0 + 2 * lh + e;
              float wkv = 0.f;
              if (n * CC + i > 0) wkv = rkc[((long)i - 1) * DD + p];
              a[e]  = wkv * sGpow[CC - 1 - i];
              bf[e] = sVnew[i][nt * 16 + ll];
            }
            acc = wmma_f32_k4(a, bf, acc);
          }
#pragma unroll
          for (int r = 0; r < 8; ++r) {
            int prow = wave * 32 + mt * 16 + r + (lh << 3);
            int col = nt * 16 + ll;
            sS[prow][col] = gC * sS[prow][col] + acc[r];
          }
        }
      }
    }
    __syncthreads();
  }
}

// ---------------------------------------------------------------------------
// Workspace layout (floats): rk | v | wkcum | o | beta   (~269 MB)
// ---------------------------------------------------------------------------
extern "C" void kernel_launch(void* const* d_in, const int* in_sizes, int n_in,
                              void* d_out, int out_size, void* d_ws,
                              size_t ws_size, hipStream_t stream) {
  const float* X   = (const float*)d_in[0];   // out (B,T,D)
  const float* Ww  = (const float*)d_in[1];   // W_write (D,D)
  const float* Wr  = (const float*)d_in[2];   // W_read  (D,D)
  const float* Wb  = (const float*)d_in[3];   // W_beta  (H,D)
  const float* dec = (const float*)d_in[4];   // decay   (H,)
  float* out = (float*)d_out;

  float* ws    = (float*)d_ws;
  float* rk    = ws;
  float* vbuf  = ws + (size_t)1 * BT * DD;
  float* wkcum = ws + (size_t)2 * BT * DD;
  float* obuf  = ws + (size_t)3 * BT * DD;
  float* beta  = ws + (size_t)4 * BT * DD;

  dim3 ggrid(DD / 128, BT / 128);   // (8, 128)
  gemm_nt_bf16_kernel<<<ggrid, 256, 0, stream>>>(X, Ww, nullptr, vbuf);
  rk_beta_kernel<<<BT, 256, 0, stream>>>(X, Wb, rk, beta);
  chunk_ut_kernel<<<BB * HH * NN, 128, 0, stream>>>(rk, beta, dec, vbuf, wkcum);
  scan_kernel<<<BB * HH * 4, 128, 0, stream>>>(rk, vbuf, wkcum, dec, obuf);
  gemm_nt_bf16_kernel<<<ggrid, 256, 0, stream>>>(obuf, Wr, X, out);
}
